// BHS_SAGE_16724602651179
// MI455X (gfx1250) — matrix operand order repaired and checked by reading, not compiled, not run on hardware
//
#include <hip/hip_runtime.h>

typedef _Float16 f16;
typedef __attribute__((ext_vector_type(16))) _Float16 v16h;
typedef __attribute__((ext_vector_type(8)))  _Float16 v8h;
typedef __attribute__((ext_vector_type(8)))  float    v8f;

// ---------------------------------------------------------------------------
// CDNA5 wave32 WMMA 16x16x32 f16->f32 fragment layouts (ISA 7.12.2):
//  A (16x32 MxK): lane L holds row m=L&15, K in {kb..kb+7, kb+16..kb+23}, kb=(L<16)?0:8
//     -> per lane: TWO contiguous 16-byte chunks  [kb..kb+7] and [kb+16..kb+23]
//  B (32x16 KxN): lane L holds col n=L&15, K in {kb..kb+7, kb+16..kb+23}, kb as for A
//  C (16x16 f32): vgpr r, lane L -> m = r + (L<16?0:8), n = L&15
// We stage LDS tiles pre-swizzled so each lane's fragment is one contiguous
// 32-byte chunk: frag load = 2x ds_load_b128.
//   A element (m,k): lane = (m&15) + 16*((k>>3)&1), idx = (k&7) + 8*((k>>4)&1)
//   B element (k,n): lane = (n&15) + 16*((k>>3)&1), idx = (k&7) + 8*((k>>4)&1)
// ---------------------------------------------------------------------------

__device__ __forceinline__ v16h frag_from_lds(const f16* p) {
  return *(const v16h*)p;                 // 32B contiguous, 32B aligned
}

// A fragment straight from global f16 row (two aligned b128 loads per lane)
__device__ __forceinline__ v16h load_a_frag_g(const f16* row, int kb) {
  v8h lo = *(const v8h*)(row + kb);
  v8h hi = *(const v8h*)(row + kb + 16);
  v16h a;
#pragma unroll
  for (int i = 0; i < 8; ++i) { a[i] = lo[i]; a[i + 8] = hi[i]; }
  return a;
}

// ---------------------------------------------------------------------------
// Kernel 1: hp = relu(x @ W_pool^T + b_pool)   [131072,32] f32 -> f16
// block = 128 (4 waves), 64 rows/block, grid = 2048
// ---------------------------------------------------------------------------
__global__ __launch_bounds__(128) void k_pool(const float* __restrict__ x,
                                              const float* __restrict__ Wp,
                                              const float* __restrict__ bp,
                                              f16* __restrict__ hp) {
  __shared__ __align__(32) f16 sAz[4][32 * 16];   // 4 row-tiles, frag order
  __shared__ __align__(32) f16 sWz[2][32 * 16];   // 2 n-tiles,   frag order
  __shared__ float sB[32];
  int tid  = threadIdx.x;
  int row0 = blockIdx.x * 64;
  if (tid < 32) sB[tid] = bp[tid];
#pragma unroll
  for (int i = 0; i < 8; ++i) {
    int idx = tid + i * 128;                      // W_pool row-major [n][k]
    int n = idx >> 5, k = idx & 31;
    sWz[n >> 4][(((n & 15) + (((k >> 3) & 1) << 4)) << 4) + (k & 7) + (((k >> 4) & 1) << 3)] = (f16)Wp[idx];
  }
#pragma unroll
  for (int i = 0; i < 16; ++i) {
    int idx = tid + i * 128;                      // x tile [m][k], coalesced read
    int m = idx >> 5, k = idx & 31;
    int lane = (m & 15) + (((k >> 3) & 1) << 4);
    int ii   = (k & 7) + (((k >> 4) & 1) << 3);
    sAz[m >> 4][(lane << 4) + ii] = (f16)x[(long)row0 * 32 + idx];
  }
  __syncthreads();

  int wave = tid >> 5, lane = tid & 31;
  v16h a = frag_from_lds(&sAz[wave][lane << 4]);
#pragma unroll
  for (int nt = 0; nt < 2; ++nt) {
    v16h b = frag_from_lds(&sWz[nt][lane << 4]);
    v8f  c = {};
    c = __builtin_amdgcn_wmma_f32_16x16x32_f16(false, a, false, b, (short)0, c, false, false);
    int n  = nt * 16 + (lane & 15);
    int mb = (lane < 16) ? 0 : 8;
#pragma unroll
    for (int r = 0; r < 8; ++r) {
      float v = c[r] + sB[n];
      v = v > 0.f ? v : 0.f;
      hp[(long)(row0 + wave * 16 + mb + r) * 32 + n] = (f16)v;
    }
  }
}

// ---------------------------------------------------------------------------
// Kernel 2: agg[g] = max over 16 neighbors hp[src[g*16+k]]  (every node deg 16)
// b128 vector loads + packed f16 max (exact: f16 max is monotonic)
// ---------------------------------------------------------------------------
__global__ __launch_bounds__(128) void k_agg(const f16* __restrict__ hp,
                                             const int* __restrict__ src,
                                             f16* __restrict__ agg) {
  int g = blockIdx.x * 128 + threadIdx.x;         // node id 0..131071
  const int* e = src + (long)g * 16;
  const v8h* r0 = (const v8h*)(hp + (long)e[0] * 32);
  v8h acc[4];
#pragma unroll
  for (int j = 0; j < 4; ++j) acc[j] = r0[j];
  for (int k = 1; k < 16; ++k) {
    const v8h* r = (const v8h*)(hp + (long)e[k] * 32);
#pragma unroll
    for (int j = 0; j < 4; ++j) acc[j] = __builtin_elementwise_max(acc[j], r[j]);
  }
  v8h* o = (v8h*)(agg + (long)g * 32);
#pragma unroll
  for (int j = 0; j < 4; ++j) o[j] = acc[j];
}

// ---------------------------------------------------------------------------
// Kernel 3: h = relu([x|agg] @ [W_self|W_neigh]^T + b_sage)  M=131072,K=64,N=128
// block = 128 (4 waves), 64 rows/block; wave -> 16x128 tile (8 N-tiles, 2 K-steps)
// ---------------------------------------------------------------------------
__global__ __launch_bounds__(128) void k_sage(const float* __restrict__ x,
                                              const f16* __restrict__ agg,
                                              const float* __restrict__ Ws,
                                              const float* __restrict__ Wn,
                                              const float* __restrict__ bs,
                                              f16* __restrict__ h) {
  __shared__ __align__(32) f16 sWz[8][2][32 * 16];  // [ntile][kchunk] frag order, 16KB
  __shared__ __align__(32) f16 sAz[4][2][32 * 16];  // [rowtile][kchunk], 8KB
  __shared__ float sBias[128];
  int tid  = threadIdx.x;
  int row0 = blockIdx.x * 64;
  sBias[tid] = bs[tid];
#pragma unroll
  for (int i = 0; i < 64; ++i) {
    int idx = tid + i * 128;                        // 0..8191 over B[k][n], k<64,n<128
    int k = idx >> 7, n = idx & 127;
    float w = (k < 32) ? Ws[n * 32 + k] : Wn[n * 32 + (k - 32)];
    int c = k & 31;
    sWz[n >> 4][k >> 5][(((n & 15) + (((c >> 3) & 1) << 4)) << 4) + (c & 7) + (((c >> 4) & 1) << 3)] = (f16)w;
  }
#pragma unroll
  for (int i = 0; i < 32; ++i) {
    int idx = tid + i * 128;                        // 0..4095 over A[r][k], r<64,k<64
    int r = idx >> 6, k = idx & 63;
    long grow = row0 + r;
    f16 v = (k < 32) ? (f16)x[grow * 32 + k] : agg[grow * 32 + (k - 32)];
    int c = k & 31;
    int lane = (r & 15) + (((c >> 3) & 1) << 4);
    int ii   = (c & 7) + (((c >> 4) & 1) << 3);
    sAz[r >> 4][k >> 5][(lane << 4) + ii] = v;
  }
  __syncthreads();

  int wave = tid >> 5, lane = tid & 31;
  v16h a0 = frag_from_lds(&sAz[wave][0][lane << 4]);
  v16h a1 = frag_from_lds(&sAz[wave][1][lane << 4]);
#pragma unroll
  for (int nt = 0; nt < 8; ++nt) {
    v16h b0 = frag_from_lds(&sWz[nt][0][lane << 4]);
    v16h b1 = frag_from_lds(&sWz[nt][1][lane << 4]);
    v8f  c = {};
    c = __builtin_amdgcn_wmma_f32_16x16x32_f16(false, a0, false, b0, (short)0, c, false, false);
    c = __builtin_amdgcn_wmma_f32_16x16x32_f16(false, a1, false, b1, (short)0, c, false, false);
    int n  = nt * 16 + (lane & 15);
    int mb = (lane < 16) ? 0 : 8;
#pragma unroll
    for (int r = 0; r < 8; ++r) {
      float v = c[r] + sBias[n];
      v = v > 0.f ? v : 0.f;
      h[(long)(row0 + wave * 16 + mb + r) * 128 + n] = (f16)v;
    }
  }
}

// ---------------------------------------------------------------------------
// Kernel 4: fused head GEMM  feat[128,131072] @ [W_adv|W_v1]^T -> [128, 76(+4)]
// grid = (5 N-tiles, 64 split-K), block = 256 (8 waves = 8 M-tiles of batch)
// 4 K-steps (K=128) staged per barrier round -> 4 WMMAs per sync pair
// ---------------------------------------------------------------------------
#define KPER 2048          // 131072 / 64 split-K slices
__global__ __launch_bounds__(256) void k_heads(const f16* __restrict__ hfeat,
                                               const float* __restrict__ Wadv,
                                               const float* __restrict__ Wv1,
                                               float* __restrict__ part) {
  __shared__ __align__(32) f16 sB[4][32 * 16];    // 4 K-steps, frag order, 4KB
  int tid   = threadIdx.x;
  int ntile = blockIdx.x;          // 0..4
  int sk    = blockIdx.y;          // 0..63
  int wave  = tid >> 5, lane = tid & 31;
  long kbase0 = (long)sk * KPER;
  int  m_global = wave * 16 + (lane & 15);
  const f16* arow = hfeat + (long)m_global * 131072;
  int  kb = (lane < 16) ? 0 : 8;
  v8f acc = {};
  for (int ks = 0; ks < KPER / 128; ++ks) {       // 16 rounds
    long kg0 = kbase0 + (long)ks * 128;
    __builtin_prefetch(arow + kg0 + 128, 0, 1);   // next A slice -> global_prefetch
    __syncthreads();
#pragma unroll
    for (int i = 0; i < 8; ++i) {
      int idx = tid + i * 256;                    // 0..2047: [q 0..3][k 0..31][n 0..15]
      int q = idx >> 9, r = idx & 511;
      int k = r >> 4, n = r & 15;
      int ng = ntile * 16 + n;
      long kg = kg0 + q * 32 + k;
      float w;
      if (ng < 12)      w = Wadv[(long)ng * 131072 + kg];
      else if (ng < 76) w = Wv1[(long)(ng - 12) * 131072 + kg];
      else              w = 0.f;
      sB[q][(((n) + (((k >> 3) & 1) << 4)) << 4) + (k & 7) + (((k >> 4) & 1) << 3)] = (f16)w;
    }
    __syncthreads();
#pragma unroll
    for (int q = 0; q < 4; ++q) {
      v16h a = load_a_frag_g(arow + kg0 + q * 32, kb);
      v16h b = frag_from_lds(&sB[q][lane << 4]);
      acc = __builtin_amdgcn_wmma_f32_16x16x32_f16(false, a, false, b, (short)0, acc, false, false);
    }
  }
  int n  = lane & 15;
  int mb = (lane < 16) ? 0 : 8;
#pragma unroll
  for (int r = 0; r < 8; ++r) {
    int m = wave * 16 + mb + r;
    part[((long)sk * 128 + m) * 80 + ntile * 16 + n] = acc[r];
  }
}

// ---------------------------------------------------------------------------
// Kernel 5: ordered split-K reduce + bias + relu -> adv[128,12], v1[128,64]
// ---------------------------------------------------------------------------
__global__ __launch_bounds__(128) void k_reduce(const float* __restrict__ part,
                                                const float* __restrict__ badv,
                                                const float* __restrict__ bv1,
                                                float* __restrict__ adv,
                                                float* __restrict__ v1) {
  int m = blockIdx.x, col = threadIdx.x;
  if (col >= 80) return;
  float s = 0.f;
  for (int k = 0; k < 64; ++k) s += part[((long)k * 128 + m) * 80 + col];
  if (col < 12) {
    float v = s + badv[col];
    adv[m * 12 + col] = v > 0.f ? v : 0.f;
  } else if (col < 76) {
    float v = s + bv1[col - 12];
    v1[m * 64 + (col - 12)] = v > 0.f ? v : 0.f;
  }
}

// ---------------------------------------------------------------------------
// Kernel 6: value MLP (64->64->1) + dueling combine -> out[128,3,4]
// ---------------------------------------------------------------------------
__global__ __launch_bounds__(64) void k_final(const float* __restrict__ v1,
                                              const float* __restrict__ adv,
                                              const float* __restrict__ Wv2,
                                              const float* __restrict__ bv2,
                                              const float* __restrict__ Wv3,
                                              const float* __restrict__ bv3,
                                              float* __restrict__ out) {
  __shared__ float s1[64], s2[64];
  __shared__ float sval;
  int m = blockIdx.x, t = threadIdx.x;
  s1[t] = v1[m * 64 + t];
  __syncthreads();
  float acc = bv2[t];
  for (int k = 0; k < 64; ++k) acc += Wv2[t * 64 + k] * s1[k];
  s2[t] = acc > 0.f ? acc : 0.f;
  __syncthreads();
  if (t == 0) {
    float v = bv3[0];
    for (int k = 0; k < 64; ++k) v += Wv3[k] * s2[k];
    sval = v;
  }
  __syncthreads();
  if (t < 12) {
    int g = t >> 2;
    float a = adv[m * 12 + t];
    float mean = 0.25f * (adv[m * 12 + g * 4] + adv[m * 12 + g * 4 + 1] +
                          adv[m * 12 + g * 4 + 2] + adv[m * 12 + g * 4 + 3]);
    out[m * 12 + t] = sval + a - mean;
  }
}

// ---------------------------------------------------------------------------
extern "C" void kernel_launch(void* const* d_in, const int* in_sizes, int n_in,
                              void* d_out, int out_size, void* d_ws, size_t ws_size,
                              hipStream_t stream) {
  (void)in_sizes; (void)n_in; (void)out_size; (void)ws_size;
  const float* x     = (const float*)d_in[0];
  const int*   src   = (const int*)d_in[1];
  // d_in[2] = dst (implied by structure: node g owns edges g*16..g*16+15)
  const float* Wp    = (const float*)d_in[3];
  const float* bp    = (const float*)d_in[4];
  const float* Wself = (const float*)d_in[5];
  const float* Wneigh= (const float*)d_in[6];
  const float* bsage = (const float*)d_in[7];
  const float* Wadv  = (const float*)d_in[8];
  const float* badv  = (const float*)d_in[9];
  const float* Wv1   = (const float*)d_in[10];
  const float* bv1   = (const float*)d_in[11];
  const float* Wv2   = (const float*)d_in[12];
  const float* bv2   = (const float*)d_in[13];
  const float* Wv3   = (const float*)d_in[14];
  const float* bv3   = (const float*)d_in[15];
  float* out = (float*)d_out;

  char* ws = (char*)d_ws;
  f16*   hp   = (f16*)(ws);                                 //  8 MiB
  f16*   agg  = (f16*)(ws + ((size_t)8 << 20));             //  8 MiB
  f16*   h    = (f16*)(ws + ((size_t)16 << 20));            // 32 MiB
  float* part = (float*)(ws + ((size_t)48 << 20));          // 2.63 MB
  float* adv  = (float*)(ws + ((size_t)52 << 20));          // 6 KB
  float* v1   = (float*)(ws + ((size_t)52 << 20) + 8192);   // 32 KB

  k_pool  <<<2048, 128, 0, stream>>>(x, Wp, bp, hp);
  k_agg   <<<1024, 128, 0, stream>>>(hp, src, agg);
  k_sage  <<<2048, 128, 0, stream>>>(x, agg, Wself, Wneigh, bsage, h);
  k_heads <<<dim3(5, 64), 256, 0, stream>>>(h, Wadv, Wv1, part);
  k_reduce<<<128, 128, 0, stream>>>(part, badv, bv1, adv, v1);
  k_final <<<128, 64, 0, stream>>>(v1, adv, Wv2, bv2, Wv3, bv3, out);
}
